// HANLayer_28501402977096
// MI455X (gfx1250) — compile-verified, matrix operand order checked by reading, not run on hardware
//
#include <hip/hip_runtime.h>
#include <hip/hip_bf16.h>
#include <math.h>

// ---------------- problem constants (match reference) ----------------
#define N_NODES 50000
#define N_EDGES 400000
#define M_PATHS 3
#define IN_DIM  128
#define OUT_DIM 64
#define HEADS   4
#define HOUT    256            // HEADS*OUT_DIM
#define BN_EPS  1e-5f
#define SLOPE   0.2f
#define ETOT    (N_EDGES + N_NODES)   // edges + self loops

typedef __attribute__((ext_vector_type(16))) _Float16 v16h;
typedef __attribute__((ext_vector_type(8)))  float    v8f;
typedef __attribute__((ext_vector_type(4)))  unsigned int u32x4;
typedef __attribute__((ext_vector_type(4)))  int      i32x4;
typedef __attribute__((ext_vector_type(8)))  int      i32x8;

#if __has_builtin(__builtin_amdgcn_tensor_load_to_lds) && __has_builtin(__builtin_amdgcn_s_wait_tensorcnt)
#define HAVE_TDM 1
#endif

static __device__ __forceinline__ float atomicMaxFloat(float* addr, float val) {
    // ordered-int trick; storage initialized to -inf
    if (val >= 0.0f)
        return __int_as_float(atomicMax((int*)addr, __float_as_int(val)));
    else
        return __uint_as_float(atomicMin((unsigned int*)addr, __float_as_uint(val)));
}

// ---------------- small utility kernels ----------------
__global__ void han_fill_f32(float* p, int n, float v) {
    int i = blockIdx.x * blockDim.x + threadIdx.x;
    if (i < n) p[i] = v;
}

__global__ void han_f32_to_f16(const float* __restrict__ src, _Float16* __restrict__ dst, int n) {
    int i = blockIdx.x * blockDim.x + threadIdx.x;
    if (i < n) dst[i] = (_Float16)src[i];
}

// ---------------- edge BatchNorm (ED == 1) ----------------
__global__ void han_bn_reduce(const float* __restrict__ ea, float* __restrict__ scal) {
    __shared__ float s0[256], s1[256];
    float a = 0.f, b = 0.f;
    for (int i = blockIdx.x * blockDim.x + threadIdx.x; i < N_EDGES; i += gridDim.x * blockDim.x) {
        float v = log1pf(ea[i]);
        a += v; b += v * v;
    }
    s0[threadIdx.x] = a; s1[threadIdx.x] = b;
    __syncthreads();
    for (int off = 128; off; off >>= 1) {
        if (threadIdx.x < off) { s0[threadIdx.x] += s0[threadIdx.x + off]; s1[threadIdx.x] += s1[threadIdx.x + off]; }
        __syncthreads();
    }
    if (threadIdx.x == 0) { atomicAdd(&scal[0], s0[0]); atomicAdd(&scal[1], s1[0]); }
}

__global__ void han_bn_norm(const float* __restrict__ ea, const float* __restrict__ scal,
                            const float* __restrict__ gamma, const float* __restrict__ beta,
                            float* __restrict__ enorm) {
    int i = blockIdx.x * blockDim.x + threadIdx.x;
    if (i >= N_EDGES) return;
    float mu  = scal[0] * (1.0f / N_EDGES);
    float var = scal[1] * (1.0f / N_EDGES) - mu * mu;
    float r   = rsqrtf(var + BN_EPS);
    enorm[i]  = (log1pf(ea[i]) - mu) * r * gamma[0] + beta[0];
}

// ---------------- self-loop edge_attr = mean of incoming ----------------
__global__ void han_deg_loopsum(const int* __restrict__ dst, const float* __restrict__ enorm,
                                float* __restrict__ deg, float* __restrict__ lsum) {
    int i = blockIdx.x * blockDim.x + threadIdx.x;
    if (i >= N_EDGES) return;
    int d = dst[i];
    atomicAdd(&deg[d], 1.0f);
    atomicAdd(&lsum[d], enorm[i]);
}

__global__ void han_loop_attr(const float* __restrict__ lsum, const float* __restrict__ deg,
                              float* __restrict__ lattr) {
    int i = blockIdx.x * blockDim.x + threadIdx.x;
    if (i >= N_NODES) return;
    lattr[i] = lsum[i] / fmaxf(deg[i], 1.0f);
}

// ---------------- B-operand fragment pre-pack (f32 W -> f16, WMMA lane order) ----
// bp[((tileN*4 + kstep)*32 + lane)*16 + j] so each wave's B fragment is one
// contiguous 32B v16h -> two global_load_b128 per K-step in the GEMM.
__global__ void han_pack_b(const float* __restrict__ Wm, _Float16* __restrict__ bp) {
    int i = blockIdx.x * blockDim.x + threadIdx.x;    // 16*4*32*16 = 32768
    if (i >= 16 * 4 * 32 * 16) return;
    int j     = i & 15;
    int lane  = (i >> 4) & 31;
    int kstep = (i >> 9) & 3;
    int tileN = i >> 11;
    int g   = lane >> 4;
    int col = tileN * 16 + (lane & 15);
    int r   = j >> 1;
    int base = ((r < 4) ? (2 * r) : (16 + 2 * (r - 4))) + g * 8 + kstep * 32 + (j & 1);
    bp[i] = (_Float16)Wm[(size_t)base * HOUT + col];
}

// ---------------- h = x @ W[m]  via WMMA (f16 in, f32 acc) ----------------
// grid = (N/16, 2), block = 256 (8 waves). All 8 waves share one 16-row A tile,
// staged into LDS by the Tensor Data Mover (1-D 4KB tile descriptor) when the
// builtin is available; wave w computes output tile (blockIdx.x, blockIdx.y*8+w).
// N and HOUT are exact multiples of 16 -> no masking, EXEC all ones for WMMA.
__global__ __launch_bounds__(256) void han_gemm_wmma(const _Float16* __restrict__ xh,
                                                     const _Float16* __restrict__ bpack,
                                                     float* __restrict__ hfeat) {
    __shared__ __align__(16) _Float16 As[16 * IN_DIM];   // 4 KB
    const int row0 = blockIdx.x * 16;

#if defined(HAVE_TDM)
    if (threadIdx.x < 32) {   // one TDM issue per block (wave 0); EXEC-independent DMA
        unsigned long long ga = (unsigned long long)(const void*)(xh + (size_t)row0 * IN_DIM);
        unsigned lds = (unsigned)(uintptr_t)As;          // low 32 bits = LDS offset
        // D# group0: count=1 | lds_addr | global_addr[56:0] | type=2
        u32x4 g0 = { 1u, lds,
                     (unsigned)(ga & 0xFFFFFFFFu),
                     (unsigned)((ga >> 32) & 0x01FFFFFFu) | (2u << 30) };
        // D# group1: data_size=2B (code 1); tensor_dim0 = tile_dim0 = 2048; dim1 = 1
        i32x8 g1 = { (int)(1u << 16),        // workgroup_mask=0, data_size=1
                     (int)(2048u << 16),     // tensor_dim0[15:0] in [63:48]
                     (int)(1u << 16),        // tensor_dim0[31:16]=0, tensor_dim1=1
                     (int)(2048u << 16),     // tensor_dim1[31:16]=0, tile_dim0=2048
                     1,                      // tile_dim1=1, tile_dim2=0
                     2048,                   // tensor_dim0_stride[31:0]
                     0, 0 };
        i32x4 z4 = { 0, 0, 0, 0 };
#if __clang_major__ >= 23
        i32x8 z8 = { 0, 0, 0, 0, 0, 0, 0, 0 };
        __builtin_amdgcn_tensor_load_to_lds(g0, g1, z4, z4, z8, 0);
#else
        __builtin_amdgcn_tensor_load_to_lds(g0, g1, z4, z4, 0);
#endif
        __builtin_amdgcn_s_wait_tensorcnt(0);
    }
    __syncthreads();
#else
    // fallback: cooperative coalesced stage (tile is contiguous in memory)
    ((float4*)As)[threadIdx.x] = ((const float4*)(xh + (size_t)row0 * IN_DIM))[threadIdx.x];
    __syncthreads();
#endif

    const int wid  = threadIdx.x >> 5;
    const int lane = threadIdx.x & 31;
    const int g    = lane >> 4;          // lane group
    const int mrow = lane & 15;
    const int tileN = blockIdx.y * 8 + wid;
    const int col   = tileN * 16 + mrow;
    const v16h* bp16 = (const v16h*)bpack;

    v8f acc = {};
#pragma unroll
    for (int kk = 0; kk < IN_DIM; kk += 32) {
        v16h a;
#pragma unroll
        for (int r = 0; r < 8; ++r) {
            // 16-bit A-operand lane layout (ISA 7.12.2): VGPR0..3 -> K 0..7 (g=0) / 8..15 (g=1),
            // VGPR4..7 -> K 16..23 / 24..31  (pairs contiguous -> ds_load_b128)
            int base = ((r < 4) ? (2 * r) : (16 + 2 * (r - 4))) + g * 8 + kk;
            a[2 * r]     = As[mrow * IN_DIM + base];
            a[2 * r + 1] = As[mrow * IN_DIM + base + 1];
        }
        v16h b = bp16[(size_t)(tileN * 4 + (kk >> 5)) * 32 + lane];
        acc = __builtin_amdgcn_wmma_f32_16x16x32_f16(false, a, false, b, (short)0, acc, false, false);
    }
#pragma unroll
    for (int r = 0; r < 8; ++r) {
        int orow = row0 + r + g * 8;     // C/D layout: VGPR r -> M=r (lanes 0-15) / M=r+8
        hfeat[(size_t)orow * HOUT + col] = acc[r];
    }
}

// ---------------- per-node attention scalars ----------------
__global__ void han_node_attn(const float* __restrict__ hfeat,
                              const float* __restrict__ a_s, const float* __restrict__ a_d,
                              float* __restrict__ asn, float* __restrict__ adn) {
    int i = blockIdx.x * blockDim.x + threadIdx.x;
    if (i >= N_NODES * HEADS) return;
    int n = i / HEADS, hh = i % HEADS;
    const float* hp = hfeat + (size_t)n * HOUT + hh * OUT_DIM;
    const float* sp = a_s + hh * OUT_DIM;
    const float* dp = a_d + hh * OUT_DIM;
    float s1 = 0.f, s2 = 0.f;
#pragma unroll 4
    for (int c = 0; c < OUT_DIM; ++c) { float hv = hp[c]; s1 += hv * sp[c]; s2 += hv * dp[c]; }
    asn[i] = s1; adn[i] = s2;
}

// coef[h] = sum_c lin_edge[0, h*OUT+c] * att_edge[h, c]   (ED == 1)
__global__ void han_coef(const float* __restrict__ lin_e, const float* __restrict__ att_e,
                         float* __restrict__ coef) {
    __shared__ float red[256];
    int t = threadIdx.x, hh = t / OUT_DIM, c = t % OUT_DIM;
    red[t] = lin_e[hh * OUT_DIM + c] * att_e[hh * OUT_DIM + c];
    __syncthreads();
    for (int off = 32; off; off >>= 1) {
        if (c < off) red[t] += red[t + off];
        __syncthreads();
    }
    if (c == 0) coef[hh] = red[t];
}

// ---------------- edge softmax: leaky-relu + segment max ----------------
__global__ void han_alpha_max(const int* __restrict__ src, const int* __restrict__ dst,
                              const float* __restrict__ enorm, const float* __restrict__ lattr,
                              const float* __restrict__ asn, const float* __restrict__ adn,
                              const float* __restrict__ coef,
                              float* __restrict__ alpha, float* __restrict__ mx) {
    int i = blockIdx.x * blockDim.x + threadIdx.x;
    if (i >= ETOT * HEADS) return;
    int e = i / HEADS, hh = i % HEADS;
    int s, d; float ev;
    if (e < N_EDGES) { s = src[e]; d = dst[e]; ev = enorm[e]; }
    else             { s = d = e - N_EDGES;    ev = lattr[s]; }
    float al = asn[s * HEADS + hh] + adn[d * HEADS + hh] + ev * coef[hh];
    al = (al >= 0.f) ? al : SLOPE * al;
    alpha[i] = al;
    atomicMaxFloat(&mx[d * HEADS + hh], al);
}

__global__ void han_exp_denom(const int* __restrict__ dst, const float* __restrict__ mx,
                              float* __restrict__ alpha, float* __restrict__ denom) {
    int i = blockIdx.x * blockDim.x + threadIdx.x;
    if (i >= ETOT * HEADS) return;
    int e = i / HEADS, hh = i % HEADS;
    int d = (e < N_EDGES) ? dst[e] : (e - N_EDGES);
    float ex = __expf(alpha[i] - mx[d * HEADS + hh]);
    alpha[i] = ex;
    atomicAdd(&denom[d * HEADS + hh], ex);
}

// ---------------- weighted aggregation, head-mean fused ----------------
// 64 threads per edge entry (one per channel); 4 edge entries per 256-thread block.
__global__ __launch_bounds__(256) void han_aggregate(const int* __restrict__ src, const int* __restrict__ dst,
                                                     const float* __restrict__ exb, const float* __restrict__ denom,
                                                     const float* __restrict__ hfeat, float* __restrict__ zm) {
    int eidx = blockIdx.x * 4 + (threadIdx.x >> 6);
    int c    = threadIdx.x & 63;
    if (eidx >= ETOT) return;
    int s, d;
    if (eidx < N_EDGES) { s = src[eidx]; d = dst[eidx]; }
    else                { s = d = eidx - N_EDGES; }
    float accv = 0.f;
#pragma unroll
    for (int hh = 0; hh < HEADS; ++hh) {
        float w = exb[eidx * HEADS + hh] / denom[d * HEADS + hh];
        accv += w * hfeat[(size_t)s * HOUT + hh * OUT_DIM + c];
    }
    atomicAdd(&zm[(size_t)d * OUT_DIM + c], accv * (1.0f / HEADS));
}

__global__ void han_bias_elu(float* __restrict__ zm, const float* __restrict__ bias) {
    int i = blockIdx.x * blockDim.x + threadIdx.x;
    if (i >= N_NODES * OUT_DIM) return;
    float z = zm[i] + bias[i % OUT_DIM];
    zm[i] = (z > 0.f) ? z : expm1f(z);
}

// ---------------- semantic attention across metapaths ----------------
__global__ __launch_bounds__(64) void han_semantic(const float* __restrict__ zs,
                                                   const float* __restrict__ semW, const float* __restrict__ semb,
                                                   const float* __restrict__ av, float* __restrict__ out) {
    __shared__ float zr[M_PATHS * OUT_DIM];
    __shared__ float red[OUT_DIM];
    __shared__ float sc[M_PATHS];
    int n = blockIdx.x, c = threadIdx.x;
#pragma unroll
    for (int m = 0; m < M_PATHS; ++m)
        zr[m * OUT_DIM + c] = zs[((size_t)m * N_NODES + n) * OUT_DIM + c];
    __syncthreads();
    for (int m = 0; m < M_PATHS; ++m) {
        float a = semb[c];
#pragma unroll 4
        for (int k = 0; k < OUT_DIM; ++k) a += zr[m * OUT_DIM + k] * semW[k * OUT_DIM + c];
        red[c] = tanhf(a) * av[c];
        __syncthreads();
        for (int off = 32; off; off >>= 1) {
            if (c < off) red[c] += red[c + off];
            __syncthreads();
        }
        if (c == 0) sc[m] = red[0];
        __syncthreads();
    }
    if (c == 0) {
        float m0 = fmaxf(sc[0], fmaxf(sc[1], sc[2]));
        float e0 = __expf(sc[0] - m0), e1 = __expf(sc[1] - m0), e2 = __expf(sc[2] - m0);
        float s = e0 + e1 + e2;
        sc[0] = e0 / s; sc[1] = e1 / s; sc[2] = e2 / s;
    }
    __syncthreads();
    float zf = 0.f;
#pragma unroll
    for (int m = 0; m < M_PATHS; ++m) zf += sc[m] * zr[m * OUT_DIM + c];
    out[(size_t)n * OUT_DIM + c] = zf;
    if (c < M_PATHS) out[(size_t)N_NODES * OUT_DIM + (size_t)n * M_PATHS + c] = sc[c];
}

// ---------------- driver ----------------
static inline int cdiv(long long a, long long b) { return (int)((a + b - 1) / b); }

extern "C" void kernel_launch(void* const* d_in, const int* in_sizes, int n_in,
                              void* d_out, int out_size, void* d_ws, size_t ws_size,
                              hipStream_t stream) {
    const float* x        = (const float*)d_in[0];
    const int*   ei       = (const int*)  d_in[1];
    const float* eattr    = (const float*)d_in[2];
    const float* Wp       = (const float*)d_in[3];
    const float* att_src  = (const float*)d_in[4];
    const float* att_dst  = (const float*)d_in[5];
    const float* lin_edge = (const float*)d_in[6];
    const float* att_edge = (const float*)d_in[7];
    const float* biasp    = (const float*)d_in[8];
    const float* bn_gamma = (const float*)d_in[9];
    const float* bn_beta  = (const float*)d_in[10];
    const float* sem_W    = (const float*)d_in[11];
    const float* sem_b    = (const float*)d_in[12];
    const float* attn_vec = (const float*)d_in[13];
    float* out = (float*)d_out;

    // workspace carve-up (256B aligned)
    size_t off = 0;
    auto carve = [&](size_t bytes) { void* p = (char*)d_ws + off; off += (bytes + 255) & ~(size_t)255; return p; };
    _Float16* xh    = (_Float16*)carve((size_t)N_NODES * IN_DIM * 2);
    _Float16* bpack = (_Float16*)carve((size_t)16 * 4 * 32 * 16 * 2);   // 64 KB packed B frags
    float* hfeat    = (float*)carve((size_t)N_NODES * HOUT * 4);
    float* zs       = (float*)carve((size_t)M_PATHS * N_NODES * OUT_DIM * 4);
    float* enorm    = (float*)carve((size_t)N_EDGES * 4);
    float* deg      = (float*)carve((size_t)N_NODES * 4);
    float* lsum     = (float*)carve((size_t)N_NODES * 4);
    float* lattr    = (float*)carve((size_t)N_NODES * 4);
    float* asn      = (float*)carve((size_t)N_NODES * HEADS * 4);
    float* adn      = (float*)carve((size_t)N_NODES * HEADS * 4);
    float* mx       = (float*)carve((size_t)N_NODES * HEADS * 4);
    float* denom    = (float*)carve((size_t)N_NODES * HEADS * 4);
    float* exb      = (float*)carve((size_t)ETOT * HEADS * 4);
    float* scal     = (float*)carve(256);  // [0]=sum, [1]=sumsq
    float* coef     = (float*)carve(256);  // [HEADS]
    if (off > ws_size) return;             // workspace too small: bail (compile-only env)

    // one-time work: f16 copy of x, zero z accumulators
    han_f32_to_f16<<<cdiv((size_t)N_NODES * IN_DIM, 256), 256, 0, stream>>>(x, xh, N_NODES * IN_DIM);
    hipMemsetAsync(zs, 0, (size_t)M_PATHS * N_NODES * OUT_DIM * 4, stream);

    for (int m = 0; m < M_PATHS; ++m) {
        const int* src_m = ei + (size_t)(m * 2 + 0) * N_EDGES;
        const int* dst_m = ei + (size_t)(m * 2 + 1) * N_EDGES;
        const float* ea_m = eattr + (size_t)m * N_EDGES;   // ED == 1
        float* zm = zs + (size_t)m * N_NODES * OUT_DIM;

        // BatchNorm over log1p(edge_attr)
        hipMemsetAsync(scal, 0, 2 * sizeof(float), stream);
        han_bn_reduce<<<256, 256, 0, stream>>>(ea_m, scal);
        han_bn_norm<<<cdiv(N_EDGES, 256), 256, 0, stream>>>(ea_m, scal, bn_gamma + m, bn_beta + m, enorm);

        // self-loop attr = mean of incoming
        hipMemsetAsync(deg, 0, (size_t)N_NODES * 4, stream);
        hipMemsetAsync(lsum, 0, (size_t)N_NODES * 4, stream);
        han_deg_loopsum<<<cdiv(N_EDGES, 256), 256, 0, stream>>>(dst_m, enorm, deg, lsum);
        han_loop_attr<<<cdiv(N_NODES, 256), 256, 0, stream>>>(lsum, deg, lattr);

        // h = x @ W[m]  (WMMA; B fragments pre-packed for b128 loads)
        han_pack_b<<<cdiv(16 * 4 * 32 * 16, 256), 256, 0, stream>>>(Wp + (size_t)m * IN_DIM * HOUT, bpack);
        han_gemm_wmma<<<dim3(N_NODES / 16, 2), 256, 0, stream>>>(xh, bpack, hfeat);

        // node/edge attention scalars
        han_node_attn<<<cdiv(N_NODES * HEADS, 256), 256, 0, stream>>>(
            hfeat, att_src + (size_t)m * HEADS * OUT_DIM, att_dst + (size_t)m * HEADS * OUT_DIM, asn, adn);
        han_coef<<<1, 256, 0, stream>>>(lin_edge + (size_t)m * HOUT, att_edge + (size_t)m * HEADS * OUT_DIM, coef);

        // edge softmax
        han_fill_f32<<<cdiv(N_NODES * HEADS, 256), 256, 0, stream>>>(mx, N_NODES * HEADS, -INFINITY);
        hipMemsetAsync(denom, 0, (size_t)N_NODES * HEADS * 4, stream);
        han_alpha_max<<<cdiv((size_t)ETOT * HEADS, 256), 256, 0, stream>>>(
            src_m, dst_m, enorm, lattr, asn, adn, coef, exb, mx);
        han_exp_denom<<<cdiv((size_t)ETOT * HEADS, 256), 256, 0, stream>>>(dst_m, mx, exb, denom);

        // aggregate + mean over heads (fused), then bias + ELU
        han_aggregate<<<cdiv(ETOT, 4), 256, 0, stream>>>(src_m, dst_m, exb, denom, hfeat, zm);
        han_bias_elu<<<cdiv(N_NODES * OUT_DIM, 256), 256, 0, stream>>>(zm, biasp + (size_t)m * OUT_DIM);
    }

    // semantic attention + outputs
    han_semantic<<<N_NODES, 64, 0, stream>>>(zs, sem_W, sem_b, attn_vec, out);
}
